// MultiHeadAttention_35072702939380
// MI455X (gfx1250) — matrix operand order repaired and checked by reading, not compiled, not run on hardware
//
#include <hip/hip_runtime.h>
#include <hip/hip_bf16.h>
#include <math.h>

// ---------------------------------------------------------------------------
// MHA for MI455X (gfx1250, wave32, WMMA, async-to-LDS, TDM).
//   B=2, S=2048, D=1024, H=16, DK=64. fp32 I/O, bf16 WMMA compute, fp32 acc.
//   Intermediates (Q,K,V,attn) kept as bf16 in workspace. Attention Q/K tiles
//   move via the Tensor Data Mover (tensor_load_to_lds, TENSORcnt); GEMM bf16
//   A-tiles move via GLOBAL_LOAD_ASYNC_TO_LDS_B128 (ASYNCcnt).
// ---------------------------------------------------------------------------

#define BATCH 2
#define SEQ   2048
#define DMODEL 1024
#define NHEAD 16
#define DHEAD 64

typedef __attribute__((ext_vector_type(16))) __bf16 v16bf;
typedef __attribute__((ext_vector_type(8)))  float  v8f;
typedef __attribute__((ext_vector_type(4)))  unsigned int u32x4;
typedef __attribute__((ext_vector_type(8)))  int i32x8;
typedef __attribute__((ext_vector_type(4)))  int i32x4;
typedef unsigned short bf16_t;

union BFrag {          // 16 bf16 = 32 bytes = two b128 LDS loads
    v16bf v;
    uint4 u4[2];
};

template <class A, class B> struct is_same_t { static constexpr bool v = false; };
template <class A> struct is_same_t<A, A>    { static constexpr bool v = true;  };

__device__ __forceinline__ bf16_t bfbits(float f) {
    union { __bf16 h; unsigned short u; } x;
    x.h = (__bf16)f;                       // hardware f32->bf16 cvt
    return x.u;
}
__device__ __forceinline__ unsigned pk2bf(float a, float b) {
    return (unsigned)bfbits(a) | ((unsigned)bfbits(b) << 16);
}

// ---- CDNA5 async copy: 16B global -> LDS, tracked by ASYNCcnt ------------
__device__ __forceinline__ void async_cp16(const bf16_t* lds_dst, const void* gsrc) {
    unsigned lds_off = (unsigned)(uintptr_t)lds_dst;   // low 32b of shared aperture = LDS offset
    asm volatile("global_load_async_to_lds_b128 %0, %1, off"
                 :: "v"(lds_off), "v"((unsigned long long)(uintptr_t)gsrc)
                 : "memory");
}
__device__ __forceinline__ void async_wait0() {
    asm volatile("s_wait_asynccnt 0x0" ::: "memory");
}

// ---- CDNA5 Tensor Data Mover: 64x64 bf16 tile, global -> LDS -------------
// Rows are 128B in global (stride DMODEL elems) and land at 144B stride in
// LDS via D# padding (pad_interval code 4 = 32 DWORDs, pad_amount code 3 =
// 4 DWORDs = 16B) so WMMA fragment addressing (LDT=72 elems) is unchanged.
__device__ __forceinline__ void tdm_load_tile_64x64(const bf16_t* lds_dst,
                                                    const bf16_t* gsrc) {
    const unsigned long long ga = (unsigned long long)(uintptr_t)gsrc;
    u32x4 g0;
    g0.x = 1u;                                            // count=1 (valid), user mode
    g0.y = (unsigned)(uintptr_t)lds_dst;                  // lds_addr (bytes)
    g0.z = (unsigned)(ga & 0xffffffffu);                  // global_addr[31:0]
    g0.w = (unsigned)((ga >> 32) & 0x01ffffffu)           // global_addr[56:32]
         | (2u << 30);                                    // type = 2 ("image")
    i32x8 g1;
    g1[0] = (1 << 16)        // data_size = 2 bytes
          | (1 << 20)        // pad_enable
          | (4 << 22)        // pad_interval: 32 DWORDs (128B) between pads
          | (3 << 25);       // pad_amount:   4 DWORDs (16B)
    g1[1] = (int)(1024u << 16);   // tensor_dim0 = 1024 elems (bits 79:48, lo16)
    g1[2] = (int)(4096u << 16);   // tensor_dim0 hi=0 | tensor_dim1 lo16 = 4096
    g1[3] = (int)(64u << 16);     // tensor_dim1 hi=0 | tile_dim0 = 64
    g1[4] = 64;                   // tile_dim1 = 64, tile_dim2 = 0
    g1[5] = 1024;                 // tensor_dim0_stride = 1024 elems (lo32)
    g1[6] = 0;                    // stride hi | tensor_dim1_stride (unused, 2D)
    g1[7] = 0;
    const i32x4 z4 = {0, 0, 0, 0};                        // groups 2/3: 2D tensor
    const i32x8 z8 = {0, 0, 0, 0, 0, 0, 0, 0};
    __builtin_amdgcn_tensor_load_to_lds(g0, g1, z4, z4, z8, 0);
}

// ---------------------------------------------------------------------------
// GEMM:  Y[M,N] = X[M,K] @ W[N,K]^T + bias[N]      (torch Linear convention)
// Block tile 128x128, K-stage 64, 256 threads (8 waves), wave tile 32x64.
// TIN:  float (convert during staging) or bf16_t (async copy to LDS).
// TOUT: float or bf16_t.
// ---------------------------------------------------------------------------
#define BM 128
#define BN 128
#define BK 64
#define LDT (BK + 8)          // bf16 elements per LDS row (144B, 16B-aligned)

template <class TIN, class TOUT>
__global__ __launch_bounds__(256)
void gemm_bias_kernel(const TIN* __restrict__ X, const float* __restrict__ W,
                      const float* __restrict__ bias, TOUT* __restrict__ Y,
                      int M, int N, int K)
{
    __shared__ bf16_t As[BM * LDT];
    __shared__ bf16_t Bs[BN * LDT];

    const int t    = threadIdx.x;
    const int lane = t & 31;
    const int wid  = t >> 5;
    const int waveM = wid & 3;        // 4 waves along M
    const int waveN = wid >> 2;       // 2 waves along N
    const int m0 = blockIdx.y * BM;
    const int n0 = blockIdx.x * BN;

    const int r = lane & 15;          // row (A) / col (B,C) within 16-tile
    const int g = lane >> 4;          // lane-half selector

    v8f acc[2][4];
#pragma unroll
    for (int i = 0; i < 2; ++i)
#pragma unroll
        for (int j = 0; j < 4; ++j) acc[i][j] = {};

    const int ldrow  = t >> 1;          // 0..127
    const int ldcol  = (t & 1) * 32;    // 0 or 32

    for (int k0 = 0; k0 < K; k0 += BK) {
        // ---- stage A tile (128x64) ----
        if constexpr (is_same_t<TIN, float>::v) {
            const float4* src = reinterpret_cast<const float4*>(
                X + (size_t)(m0 + ldrow) * K + k0 + ldcol);
            bf16_t* dst = &As[ldrow * LDT + ldcol];
#pragma unroll
            for (int i = 0; i < 8; ++i) {
                float4 v = src[i];
                uint2 o; o.x = pk2bf(v.x, v.y); o.y = pk2bf(v.z, v.w);
                *reinterpret_cast<uint2*>(&dst[4 * i]) = o;
            }
            if (k0 + BK < K)
                __builtin_prefetch(X + (size_t)(m0 + ldrow) * K + k0 + BK + ldcol, 0, 1);
        } else {
            // bf16 input: pure copy -> async DMA to LDS (ASYNCcnt path)
            const bf16_t* gsrc = (const bf16_t*)X + (size_t)(m0 + ldrow) * K + k0 + ldcol;
            const bf16_t* ldst = &As[ldrow * LDT + ldcol];
#pragma unroll
            for (int i = 0; i < 4; ++i)
                async_cp16(ldst + i * 8, gsrc + i * 8);
        }
        // ---- stage B tile (weights fp32 -> bf16) ----
        {
            const float4* src = reinterpret_cast<const float4*>(
                W + (size_t)(n0 + ldrow) * K + k0 + ldcol);
            bf16_t* dst = &Bs[ldrow * LDT + ldcol];
#pragma unroll
            for (int i = 0; i < 8; ++i) {
                float4 v = src[i];
                uint2 o; o.x = pk2bf(v.x, v.y); o.y = pk2bf(v.z, v.w);
                *reinterpret_cast<uint2*>(&dst[4 * i]) = o;
            }
            if (k0 + BK < K)
                __builtin_prefetch(W + (size_t)(n0 + ldrow) * K + k0 + BK + ldcol, 0, 1);
        }
        if constexpr (!is_same_t<TIN, float>::v) async_wait0();
        __syncthreads();

#pragma unroll
        for (int ks = 0; ks < 2; ++ks) {            // two K=32 WMMA steps
            BFrag aF[2], bF[4];
#pragma unroll
            for (int im = 0; im < 2; ++im) {
                // A 16x32 bf16 layout: lane r = row; half g holds K {g*8..+7, +16}
                int base = (waveM * 32 + im * 16 + r) * LDT + ks * 32 + g * 8;
                aF[im].u4[0] = *reinterpret_cast<const uint4*>(&As[base]);
                aF[im].u4[1] = *reinterpret_cast<const uint4*>(&As[base + 16]);
            }
#pragma unroll
            for (int in = 0; in < 4; ++in) {
                // B 32x16 bf16 layout: lane r = col n; half g holds K {g*8..+7, +16}
                int base = (waveN * 64 + in * 16 + r) * LDT + ks * 32 + g * 8;
                bF[in].u4[0] = *reinterpret_cast<const uint4*>(&Bs[base]);
                bF[in].u4[1] = *reinterpret_cast<const uint4*>(&Bs[base + 16]);
            }
#pragma unroll
            for (int im = 0; im < 2; ++im)
#pragma unroll
                for (int in = 0; in < 4; ++in)
                    acc[im][in] = __builtin_amdgcn_wmma_f32_16x16x32_bf16(
                        false, aF[im].v, false, bF[in].v,
                        (short)0, acc[im][in], false, false);
        }
        __syncthreads();
    }

    // ---- epilogue: bias + store (C layout: elem rr -> row rr + g*8, lane r -> col) ----
#pragma unroll
    for (int in = 0; in < 4; ++in) {
        const int col = n0 + waveN * 64 + in * 16 + r;
        const float bv = bias[col];
#pragma unroll
        for (int im = 0; im < 2; ++im) {
            const int mrow = m0 + waveM * 32 + im * 16 + g * 8;
#pragma unroll
            for (int rr = 0; rr < 8; ++rr) {
                const float val = acc[im][in][rr] + bv;
                if constexpr (is_same_t<TOUT, float>::v)
                    Y[(size_t)(mrow + rr) * N + col] = val;
                else
                    Y[(size_t)(mrow + rr) * N + col] = bfbits(val);
            }
        }
    }
}

// ---------------------------------------------------------------------------
// Flash attention (causal), bf16 in / bf16 out. Grid: (S/64, B*H).
// 128 threads = 4 waves; each wave owns 16 query rows.
// Q/K tiles arrive via TDM (wave 0 issues one descriptor per tile);
// V is staged transposed for the PV B-operand.
// ---------------------------------------------------------------------------
__global__ __launch_bounds__(128)
void attn_kernel(const bf16_t* __restrict__ Q, const bf16_t* __restrict__ K,
                 const bf16_t* __restrict__ V, bf16_t* __restrict__ O)
{
    __shared__ bf16_t Qs[64 * LDT];   // [q][dk]
    __shared__ bf16_t Ks[64 * LDT];   // [key][dk]
    __shared__ bf16_t Vt[64 * LDT];   // [dk][key]  (transposed)
    __shared__ bf16_t Ps[64 * LDT];   // [wid*16+q][key] per-wave P

    const int t    = threadIdx.x;
    const int lane = t & 31;
    const int wid  = t >> 5;
    const int r    = lane & 15;
    const int g    = lane >> 4;

    const int bx   = blockIdx.x;            // query tile (64 rows)
    const int b    = blockIdx.y / NHEAD;
    const int h    = blockIdx.y % NHEAD;
    const size_t headoff = (size_t)b * SEQ * DMODEL + h * DHEAD;
    const int qblk = bx * 64;
    const float scale = 0.125f;             // 1/sqrt(64)

    const int ldrow = t >> 1;               // 0..63
    const int ldcol = (t & 1) * 32;         // bf16 elements

    // ---- stage Q tile once via TDM ----
    if (wid == 0) {
        tdm_load_tile_64x64(Qs, Q + headoff + (size_t)qblk * DMODEL);
        __builtin_amdgcn_s_wait_tensorcnt(0);
    }
    __syncthreads();

    // Q fragments (A layout), kept in registers: dk halves 0..31 / 32..63
    BFrag qf[2];
#pragma unroll
    for (int kh = 0; kh < 2; ++kh) {
        int base = (wid * 16 + r) * LDT + kh * 32 + g * 8;
        qf[kh].u4[0] = *reinterpret_cast<const uint4*>(&Qs[base]);
        qf[kh].u4[1] = *reinterpret_cast<const uint4*>(&Qs[base + 16]);
    }

    float m_i[8], l_i[8];
    v8f oacc[4];
#pragma unroll
    for (int rr = 0; rr < 8; ++rr) { m_i[rr] = -1e30f; l_i[rr] = 0.f; }
#pragma unroll
    for (int in = 0; in < 4; ++in) oacc[in] = {};

    for (int j = 0; j <= bx; ++j) {
        // ---- stage K tile via TDM (single descriptor, wave 0) ----
        if (wid == 0)
            tdm_load_tile_64x64(Ks, K + headoff + (size_t)(j * 64) * DMODEL);
        // ---- stage V tile transposed: Vt[dk][key] (all threads) ----
        {
            const uint4* src = reinterpret_cast<const uint4*>(
                V + headoff + (size_t)(j * 64 + ldrow) * DMODEL + ldcol);
#pragma unroll
            for (int i = 0; i < 4; ++i) {
                uint4 v = src[i];          // 8 bf16
                const int cb = ldcol + i * 8;
                Vt[(cb + 0) * LDT + ldrow] = (bf16_t)(v.x);
                Vt[(cb + 1) * LDT + ldrow] = (bf16_t)(v.x >> 16);
                Vt[(cb + 2) * LDT + ldrow] = (bf16_t)(v.y);
                Vt[(cb + 3) * LDT + ldrow] = (bf16_t)(v.y >> 16);
                Vt[(cb + 4) * LDT + ldrow] = (bf16_t)(v.z);
                Vt[(cb + 5) * LDT + ldrow] = (bf16_t)(v.z >> 16);
                Vt[(cb + 6) * LDT + ldrow] = (bf16_t)(v.w);
                Vt[(cb + 7) * LDT + ldrow] = (bf16_t)(v.w >> 16);
            }
        }
        if (wid == 0) __builtin_amdgcn_s_wait_tensorcnt(0);
        __syncthreads();

        // ---- scores S = Q K^T * scale  (4 key sub-tiles of 16) ----
        v8f s[4];
#pragma unroll
        for (int in = 0; in < 4; ++in) {
            s[in] = {};
#pragma unroll
            for (int kh = 0; kh < 2; ++kh) {
                BFrag kF;   // B layout: lane r = key col; half g holds dk {g*8..+7, +16}
                int base = (in * 16 + r) * LDT + kh * 32 + g * 8;
                kF.u4[0] = *reinterpret_cast<const uint4*>(&Ks[base]);
                kF.u4[1] = *reinterpret_cast<const uint4*>(&Ks[base + 16]);
                s[in] = __builtin_amdgcn_wmma_f32_16x16x32_bf16(
                    false, qf[kh].v, false, kF.v, (short)0, s[in], false, false);
            }
#pragma unroll
            for (int rr = 0; rr < 8; ++rr) s[in][rr] *= scale;
        }

        // ---- causal mask on the diagonal tile ----
        if (j == bx) {
#pragma unroll
            for (int in = 0; in < 4; ++in) {
                const int kcol = in * 16 + r;
#pragma unroll
                for (int rr = 0; rr < 8; ++rr) {
                    const int qrow = wid * 16 + rr + g * 8;
                    if (kcol > qrow) s[in][rr] = -1e30f;
                }
            }
        }

        // ---- online softmax (per row; lanes 0-15 hold rows m, 16-31 rows m+8) ----
#pragma unroll
        for (int rr = 0; rr < 8; ++rr) {
            float mv = s[0][rr];
#pragma unroll
            for (int in = 1; in < 4; ++in) mv = fmaxf(mv, s[in][rr]);
#pragma unroll
            for (int off = 1; off < 16; off <<= 1)
                mv = fmaxf(mv, __shfl_xor(mv, off, 32));
            const float newm  = fmaxf(m_i[rr], mv);
            const float alpha = __expf(m_i[rr] - newm);
            float lsum = 0.f;
#pragma unroll
            for (int in = 0; in < 4; ++in) {
                float p = __expf(s[in][rr] - newm);
                s[in][rr] = p;
                lsum += p;
            }
#pragma unroll
            for (int off = 1; off < 16; off <<= 1)
                lsum += __shfl_xor(lsum, off, 32);
            l_i[rr] = l_i[rr] * alpha + lsum;
            m_i[rr] = newm;
#pragma unroll
            for (int in = 0; in < 4; ++in) oacc[in][rr] *= alpha;
        }

        // ---- P -> bf16 into per-wave LDS (C layout -> memory [row][key]) ----
#pragma unroll
        for (int in = 0; in < 4; ++in)
#pragma unroll
            for (int rr = 0; rr < 8; ++rr)
                Ps[(wid * 16 + rr + g * 8) * LDT + in * 16 + r] = bfbits(s[in][rr]);

        // ---- O += P V   (A = P from LDS, B = Vt contiguous) ----
#pragma unroll
        for (int in = 0; in < 4; ++in) {        // dk sub-tiles
#pragma unroll
            for (int kh = 0; kh < 2; ++kh) {    // key halves 0-31 / 32-63
                BFrag pF, vF;
                int pbase = (wid * 16 + r) * LDT + kh * 32 + g * 8;
                pF.u4[0] = *reinterpret_cast<const uint4*>(&Ps[pbase]);
                pF.u4[1] = *reinterpret_cast<const uint4*>(&Ps[pbase + 16]);
                int vbase = (in * 16 + r) * LDT + kh * 32 + g * 8;
                vF.u4[0] = *reinterpret_cast<const uint4*>(&Vt[vbase]);
                vF.u4[1] = *reinterpret_cast<const uint4*>(&Vt[vbase + 16]);
                oacc[in] = __builtin_amdgcn_wmma_f32_16x16x32_bf16(
                    false, pF.v, false, vF.v, (short)0, oacc[in], false, false);
            }
        }
        __syncthreads();   // protect Ks/Vt before next iteration restages
    }

    // ---- normalize + store (bf16) ----
#pragma unroll
    for (int in = 0; in < 4; ++in) {
        const int col = h * DHEAD + in * 16 + r;
#pragma unroll
        for (int rr = 0; rr < 8; ++rr) {
            const int row = qblk + wid * 16 + rr + g * 8;
            O[(size_t)b * SEQ * DMODEL + (size_t)row * DMODEL + col] =
                bfbits(oacc[in][rr] / l_i[rr]);
        }
    }
}

// ---------------------------------------------------------------------------
// Launcher
// ---------------------------------------------------------------------------
extern "C" void kernel_launch(void* const* d_in, const int* in_sizes, int n_in,
                              void* d_out, int out_size, void* d_ws, size_t ws_size,
                              hipStream_t stream)
{
    const float* query = (const float*)d_in[0];
    const float* key   = (const float*)d_in[1];
    const float* value = (const float*)d_in[2];
    // d_in[3] = mask (tril, implemented analytically)
    const float* Wq = (const float*)d_in[4];
    const float* bq = (const float*)d_in[5];
    const float* Wk = (const float*)d_in[6];
    const float* bk = (const float*)d_in[7];
    const float* Wv = (const float*)d_in[8];
    const float* bv = (const float*)d_in[9];
    const float* Wo = (const float*)d_in[10];
    const float* bo = (const float*)d_in[11];
    float* out = (float*)d_out;

    const size_t plane = (size_t)BATCH * SEQ * DMODEL;   // 4M elements
    bf16_t* Qp = (bf16_t*)d_ws;
    bf16_t* Kp = Qp + plane;
    bf16_t* Vp = Kp + plane;
    bf16_t* Ap = Vp + plane;

    const int M = BATCH * SEQ;           // 4096
    dim3 gblk(256);
    dim3 ggrd(DMODEL / BN, M / BM);      // (8, 32)

    gemm_bias_kernel<float, bf16_t><<<ggrd, gblk, 0, stream>>>(query, Wq, bq, Qp, M, DMODEL, DMODEL);
    gemm_bias_kernel<float, bf16_t><<<ggrd, gblk, 0, stream>>>(key,   Wk, bk, Kp, M, DMODEL, DMODEL);
    gemm_bias_kernel<float, bf16_t><<<ggrd, gblk, 0, stream>>>(value, Wv, bv, Vp, M, DMODEL, DMODEL);

    dim3 ablk(128);
    dim3 agrd(SEQ / 64, BATCH * NHEAD);  // (32, 32)
    attn_kernel<<<agrd, ablk, 0, stream>>>(Qp, Kp, Vp, Ap);

    gemm_bias_kernel<bf16_t, float><<<ggrd, gblk, 0, stream>>>(Ap, Wo, bo, out, M, DMODEL, DMODEL);
}
